// CausalSelfAttention_3642132267025
// MI455X (gfx1250) — compile-verified, matrix operand order checked by reading, not run on hardware
//
#include <hip/hip_runtime.h>
#include <hip/hip_bf16.h>

// CausalSelfAttention on MI455X (gfx1250, wave32, WMMA bf16 path).
// B=128, T=512, C=512, H=16, D=32.

#define B_SZ 128
#define T_SZ 512
#define C_SZ 512
#define H_SZ 16
#define D_SZ 32

typedef __attribute__((ext_vector_type(16))) __bf16 v16bf;
typedef __attribute__((ext_vector_type(8)))  float  v8f;

union AFrag { __bf16 e[16]; v16bf v; };
union CFrag { float  e[8];  v8f   v; };

// K-index mapping for 16-bit A/B fragments (ISA 7.12.2):
// lanes 0-15: VGPR0..3 hold K=0..7, VGPR4..7 hold K=16..23
// lanes 16-31: VGPR0..3 hold K=8..15, VGPR4..7 hold K=24..31
__device__ __forceinline__ int kmap(int e, int half) {
    return (e < 8) ? (half * 8 + e) : (16 + half * 8 + (e - 8));
}

// ---- CDNA5 async global->LDS staging (ASYNCcnt path), guarded ------------
#if defined(__has_builtin)
#if __has_builtin(__builtin_amdgcn_global_load_async_to_lds_b128)
#define HAVE_ASYNC_LDS 1
#endif
#endif

#ifdef HAVE_ASYNC_LDS
// Signature (from toolchain diagnostic): param0 = v4i* in AS(1), then LDS
// pointer, imm offset, imm cpol.
typedef int v4i_t __attribute__((vector_size(16)));
typedef __attribute__((address_space(1))) v4i_t* g_v4i_p;
typedef __attribute__((address_space(3))) v4i_t* l_v4i_p;
#define ASYNC_COPY16(gp, lp)                                                   \
    __builtin_amdgcn_global_load_async_to_lds_b128(                            \
        (g_v4i_p)(v4i_t*)(gp), (l_v4i_p)(v4i_t*)(lp), 0, 0)
#if __has_builtin(__builtin_amdgcn_s_wait_asynccnt)
#define STAGE_WAIT() __builtin_amdgcn_s_wait_asynccnt(0)
#else
#define STAGE_WAIT() asm volatile("s_wait_asynccnt 0" ::: "memory")
#endif
#else
#define STAGE_WAIT()
#endif

// ---------------------------------------------------------------------------
// Tiled GEMM with bias: Out[M,N] = A[M,K] * Bw[K,N] + bias[N]
// A: f32 or bf16 (templated); Bw/bias: f32; Out: bf16 or f32 (templated).
// Block tile 128x128, BK=32, double-buffered LDS staged via async DMA.
// 8 waves, each computing 32x64 (8 WMMA accumulators).
// f32->bf16 conversion happens during fragment build from LDS.
// ---------------------------------------------------------------------------
template <typename TA, typename TO>
__global__ void __launch_bounds__(256)
gemm_bias_wmma(const TA* __restrict__ A, const float* __restrict__ Bw,
               const float* __restrict__ bias, TO* __restrict__ Out,
               int M, int N, int K) {
    constexpr int BM = 128, BN = 128, BK = 32;
    constexpr int APAD = (sizeof(TA) == 4) ? 36 : 40;  // 16B-aligned, bank-safe
    constexpr int BPAD = 132;                          // halves 32 banks apart
    __shared__ TA    AsubT[2][BM][APAD];
    __shared__ float BsubF[2][BK][BPAD];

    const int tid   = threadIdx.x;
    const int lane  = tid & 31;
    const int wid   = tid >> 5;      // 8 waves
    const int waveM = wid >> 1;      // 0..3  (M, 32 rows each)
    const int waveN = wid & 1;       // 0..1  (N, 64 cols each)
    const int half  = lane >> 4;
    const int ln16  = lane & 15;

    const int gm0 = blockIdx.y * BM;
    const int gn0 = blockIdx.x * BN;

    // Per-thread staging map: each thread moves 16 contiguous elements per tile.
    const int arow  = tid >> 1;              // 0..127
    const int akoff = (tid & 1) * 16;        // 0 or 16
    const int bkrow = tid >> 3;              // 0..31
    const int bnoff = (tid & 7) * 16;        // 0..112

    CFrag acc[2][4];
    #pragma unroll
    for (int i = 0; i < 2; ++i)
        #pragma unroll
        for (int j = 0; j < 4; ++j)
            #pragma unroll
            for (int e = 0; e < 8; ++e)
                acc[i][j].e[e] = 0.0f;

    const int nk = K / BK;

    // ---- staging one (A,B) tile pair into LDS buffer `buf` ----
    auto stage = [&](int buf, int kk) {
        const TA*    ga = A  + (size_t)(gm0 + arow) * K + kk + akoff;
        const float* gb = Bw + (size_t)(kk + bkrow) * N + gn0 + bnoff;
        TA*    la = &AsubT[buf][arow][akoff];
        float* lb = &BsubF[buf][bkrow][bnoff];
#ifdef HAVE_ASYNC_LDS
        constexpr int CHA = (16 * (int)sizeof(TA)) / 16;  // 16B chunks for A
        #pragma unroll
        for (int c = 0; c < CHA; ++c)
            ASYNC_COPY16((const char*)ga + 16 * c, (char*)la + 16 * c);
        #pragma unroll
        for (int c = 0; c < 4; ++c)                       // 16 f32 = 4 chunks
            ASYNC_COPY16((const char*)gb + 16 * c, (char*)lb + 16 * c);
#else
        #pragma unroll
        for (int i = 0; i < 16; ++i) la[i] = ga[i];
        #pragma unroll
        for (int i = 0; i < 16; ++i) lb[i] = gb[i];
#endif
    };

    // ---- one k-step of WMMA from LDS buffer `buf` ----
    auto compute = [&](int buf) {
        AFrag afr[2], bfr[4];
        #pragma unroll
        for (int t = 0; t < 2; ++t) {
            const int r = waveM * 32 + t * 16 + ln16;
            #pragma unroll
            for (int e = 0; e < 16; ++e)
                afr[t].e[e] = (__bf16)(float)AsubT[buf][r][kmap(e, half)];
        }
        #pragma unroll
        for (int t = 0; t < 4; ++t) {
            const int n = waveN * 64 + t * 16 + ln16;
            #pragma unroll
            for (int e = 0; e < 16; ++e)
                bfr[t].e[e] = (__bf16)BsubF[buf][kmap(e, half)][n];
        }
        #pragma unroll
        for (int i = 0; i < 2; ++i)
            #pragma unroll
            for (int j = 0; j < 4; ++j)
                acc[i][j].v = __builtin_amdgcn_wmma_f32_16x16x32_bf16(
                    false, afr[i].v, false, bfr[j].v,
                    (short)0, acc[i][j].v, false, false);
    };

    // ---- software pipeline: stage(i+1) overlaps compute(i) ----
    stage(0, 0);
    STAGE_WAIT();
    __syncthreads();
    for (int i = 0; i < nk; ++i) {
        if (i + 1 < nk) stage((i + 1) & 1, (i + 1) * BK);
        compute(i & 1);
        if (i + 1 < nk) { STAGE_WAIT(); }
        __syncthreads();
    }

    // Epilogue: bias + store (C layout: n = lane%16, m = vgpr + 8*half)
    #pragma unroll
    for (int i = 0; i < 2; ++i) {
        #pragma unroll
        for (int j = 0; j < 4; ++j) {
            const int gn = gn0 + waveN * 64 + j * 16 + ln16;
            const float bv = bias[gn];
            #pragma unroll
            for (int e = 0; e < 8; ++e) {
                const int gm = gm0 + waveM * 32 + i * 16 + e + 8 * half;
                Out[(size_t)gm * N + gn] = (TO)(acc[i][j].e[e] + bv);
            }
        }
    }
}

// ---------------------------------------------------------------------------
// Fused causal flash-attention. qkv: [B*T, 3C] bf16 (Q|K|V). y: [B*T, C] bf16.
// One wave per 16-query tile per (b,h). Processes 32 keys (2 tiles) per step:
//   S = Q K^T (1 WMMA per 16-key tile, K-dim = D = 32)
//   online softmax in f32, P transposed C->A layout via per-wave LDS
//   O += P V   (2 WMMAs for the two 16-wide halves of D, K-dim = 32 keys)
// ---------------------------------------------------------------------------
__global__ void __launch_bounds__(128)
attn_fused_wmma(const __bf16* __restrict__ qkv, __bf16* __restrict__ y) {
    __shared__ __bf16 Pld[4][16][34];   // per-wave P staging (16 x 32 + pad)

    const int tid  = threadIdx.x;
    const int lane = tid & 31;
    const int w    = tid >> 5;     // wave in block (0..3)
    const int half = lane >> 4;
    const int ln16 = lane & 15;

    const int gw = blockIdx.x * 4 + w;   // global q-tile id
    const int qt = gw & 31;              // query tile (T/16 = 32)
    const int h  = (gw >> 5) & 15;
    const int b  = gw >> 9;

    const size_t rowQ  = (size_t)(b * T_SZ + qt * 16);
    const float  scale = 0.17677669529663687f;   // 1/sqrt(D)

    // Q fragment, A layout (row = lane%16, K along elems)
    AFrag qa;
    {
        const size_t base = (rowQ + ln16) * (3 * C_SZ) + (size_t)h * D_SZ;
        #pragma unroll
        for (int e = 0; e < 16; ++e)
            qa.e[e] = qkv[base + kmap(e, half)];
    }

    CFrag o0, o1;
    float mrow[8], lrow[8];
    #pragma unroll
    for (int e = 0; e < 8; ++e) {
        o0.e[e] = 0.0f; o1.e[e] = 0.0f;
        mrow[e] = -3.0e38f; lrow[e] = 0.0f;
    }

    const int npairs = (qt >> 1) + 1;
    for (int p = 0; p < npairs; ++p) {
        const int j0 = 2 * p;

        // K^T fragments, B layout (n = key = lane%16, K = head dim d)
        AFrag kb0, kb1;
        {
            const size_t base0 =
                (size_t)(b * T_SZ + j0 * 16 + ln16) * (3 * C_SZ) + C_SZ + h * D_SZ;
            const size_t base1 = base0 + (size_t)16 * (3 * C_SZ);
            #pragma unroll
            for (int e = 0; e < 16; ++e) {
                const int d = kmap(e, half);
                kb0.e[e] = qkv[base0 + d];
                kb1.e[e] = qkv[base1 + d];
            }
        }

        v8f z = {};
        CFrag s0, s1;
        s0.v = __builtin_amdgcn_wmma_f32_16x16x32_bf16(
            false, qa.v, false, kb0.v, (short)0, z, false, false);
        s1.v = __builtin_amdgcn_wmma_f32_16x16x32_bf16(
            false, qa.v, false, kb1.v, (short)0, z, false, false);

        // scale + causal mask + online softmax (stats per lane cover 8 rows)
        float pv0[8], pv1[8];
        #pragma unroll
        for (int e = 0; e < 8; ++e) {
            const int qrow = qt * 16 + e + 8 * half;
            const int k0   = j0 * 16 + ln16;
            float a = (k0      <= qrow) ? s0.e[e] * scale : -3.0e38f;
            float c = (k0 + 16 <= qrow) ? s1.e[e] * scale : -3.0e38f;
            float rm = fmaxf(a, c);
            #pragma unroll
            for (int m = 1; m < 16; m <<= 1)        // stays within lane-half
                rm = fmaxf(rm, __shfl_xor(rm, m, 32));
            const float mnew  = fmaxf(mrow[e], rm);
            const float alpha = __expf(mrow[e] - mnew);
            const float p0 = __expf(a - mnew);
            const float p1 = __expf(c - mnew);
            float rs = p0 + p1;
            #pragma unroll
            for (int m = 1; m < 16; m <<= 1)
                rs += __shfl_xor(rs, m, 32);
            lrow[e] = lrow[e] * alpha + rs;
            mrow[e] = mnew;
            o0.e[e] *= alpha;
            o1.e[e] *= alpha;
            pv0[e] = p0; pv1[e] = p1;
        }

        // Transpose P from C layout to A layout through LDS (intra-wave only:
        // lockstep wave + s_wait_dscnt is sufficient, no workgroup barrier —
        // waves in this block have different causal trip counts).
        #pragma unroll
        for (int e = 0; e < 8; ++e) {
            Pld[w][e + 8 * half][ln16]      = (__bf16)pv0[e];
            Pld[w][e + 8 * half][16 + ln16] = (__bf16)pv1[e];
        }
        asm volatile("s_wait_dscnt 0" ::: "memory");
        AFrag pa;
        #pragma unroll
        for (int e = 0; e < 16; ++e)
            pa.e[e] = Pld[w][ln16][kmap(e, half)];

        // V fragments, B layout (n = dim = lane%16, K = key index 0..31)
        AFrag vb0, vb1;
        {
            const size_t vcol = (size_t)2 * C_SZ + h * D_SZ + ln16;
            const size_t rowV = (size_t)(b * T_SZ + j0 * 16);
            #pragma unroll
            for (int e = 0; e < 16; ++e) {
                const size_t r = (rowV + kmap(e, half)) * (3 * C_SZ);
                vb0.e[e] = qkv[r + vcol];
                vb1.e[e] = qkv[r + vcol + 16];
            }
        }
        o0.v = __builtin_amdgcn_wmma_f32_16x16x32_bf16(
            false, pa.v, false, vb0.v, (short)0, o0.v, false, false);
        o1.v = __builtin_amdgcn_wmma_f32_16x16x32_bf16(
            false, pa.v, false, vb1.v, (short)0, o1.v, false, false);
    }

    // Normalize and store y (C layout)
    #pragma unroll
    for (int e = 0; e < 8; ++e) {
        const float  rl  = 1.0f / lrow[e];
        const size_t row = rowQ + e + 8 * half;
        y[row * C_SZ + h * D_SZ + ln16]      = (__bf16)(o0.e[e] * rl);
        y[row * C_SZ + h * D_SZ + 16 + ln16] = (__bf16)(o1.e[e] * rl);
    }
}

// ---------------------------------------------------------------------------
extern "C" void kernel_launch(void* const* d_in, const int* in_sizes, int n_in,
                              void* d_out, int out_size, void* d_ws, size_t ws_size,
                              hipStream_t stream) {
    (void)in_sizes; (void)n_in; (void)out_size; (void)ws_size;

    const float* x      = (const float*)d_in[0];
    const float* W_attn = (const float*)d_in[1];
    const float* b_attn = (const float*)d_in[2];
    const float* W_proj = (const float*)d_in[3];
    const float* b_proj = (const float*)d_in[4];
    float* out = (float*)d_out;

    const int M = B_SZ * T_SZ;                     // 65536 rows
    __bf16* qkv = (__bf16*)d_ws;                   // [M, 3C] bf16 = 192 MiB
    __bf16* ybf = (__bf16*)((char*)d_ws +
                  (size_t)M * 3 * C_SZ * sizeof(__bf16));  // [M, C] bf16 = 64 MiB

    // 1) qkv = x @ W_attn + b_attn   (f32 in -> bf16 out)
    gemm_bias_wmma<float, __bf16>
        <<<dim3((3 * C_SZ) / 128, M / 128), 256, 0, stream>>>(
            x, W_attn, b_attn, qkv, M, 3 * C_SZ, C_SZ);

    // 2) y = causal_softmax(Q K^T / sqrt(D)) V
    attn_fused_wmma
        <<<dim3((B_SZ * H_SZ * (T_SZ / 16)) / 4), 128, 0, stream>>>(qkv, ybf);

    // 3) out = y @ W_proj + b_proj   (bf16 in -> f32 out)
    gemm_bias_wmma<__bf16, float>
        <<<dim3(C_SZ / 128, M / 128), 256, 0, stream>>>(
            ybf, W_proj, b_proj, out, M, C_SZ, C_SZ);
}
